// TreeAutoEncoder_17446157157127
// MI455X (gfx1250) — compile-verified, matrix operand order checked by reading, not run on hardware
//
#include <hip/hip_runtime.h>

// ---------------- model constants ----------------
#define Bc      64
#define Sc      64
#define Hc      256
#define NHEADc  8
#define DHc     32
#define FFc     1024
#define NLAY    3
#define DEPTHc  9
#define NPTc    511          // 2^9 - 1
#define NNc     (Bc*NPTc)    // 32704
#define LBLc    2000
#define LBLpad  2048         // logit weight rows padded for guard-free B loads
#define BSc     (Bc*Sc)      // 4096

typedef __attribute__((ext_vector_type(16))) _Float16 v16h;
typedef __attribute__((ext_vector_type(8)))  _Float16 v8h;
typedef __attribute__((ext_vector_type(8)))  float    v8f;
typedef _Float16 h16;

__device__ __forceinline__ float sigm(float x) { return 1.0f / (1.0f + expf(-x)); }

// load a 16-half fragment: two contiguous 128-bit chunks at p and p+16
__device__ __forceinline__ v16h ldfrag(const h16* __restrict__ p) {
    v8h lo = *(const v8h*)p;
    v8h hi = *(const v8h*)(p + 16);
    v16h r;
#pragma unroll
    for (int j = 0; j < 8; j++) { r[j] = lo[j]; r[8 + j] = hi[j]; }
    return r;
}

// =====================================================================
// WMMA GEMM:  C[M,N](f32) = A[M,K](f16,row-major) * Bt[Npad,K](f16 = B^T)
//             (+ bias[N]) (+ C if accFlag)
// Block = 256 threads = 8 waves arranged 2(M) x 4(N).
// Each wave computes a 32x64 tile: 2 A-frags x 4 B-frags -> 8 WMMA / K-step
// (register blocking: 1.5 b128 loads per WMMA instead of 4).
// Requirements: M % 64 == 0; Bt has ceil(N/256)*256 valid (padded) rows;
// C stores are guarded per-lane with col < N.
// =====================================================================
__global__ __launch_bounds__(256) void k_wmma_gemm(
        const h16* __restrict__ A, const h16* __restrict__ Bt,
        const float* __restrict__ bias, float* __restrict__ C,
        int M, int N, int K, int accFlag) {
    int lane = threadIdx.x & 31;
    int wave = threadIdx.x >> 5;
    int mRow = blockIdx.y * 64  + (wave >> 2) * 32;   // wave M base (32 rows)
    int nCol = blockIdx.x * 256 + (wave & 3) * 64;    // wave N base (64 cols)
    int half = lane >> 4;                             // 0 or 1
    int l15  = lane & 15;

    const h16* a0p = A  + (size_t)(mRow + l15)      * K + half * 8;
    const h16* a1p = A  + (size_t)(mRow + 16 + l15) * K + half * 8;
    const h16* bp  = Bt + (size_t)(nCol + l15)      * K + half * 8;
    const size_t bs = (size_t)16 * K;                 // B subtile stride (16 cols)

    v8f acc[8];
#pragma unroll
    for (int i = 0; i < 8; i++) acc[i] = (v8f){0.f,0.f,0.f,0.f,0.f,0.f,0.f,0.f};

    for (int k0 = 0; k0 < K; k0 += 32) {
        v16h a0 = ldfrag(a0p + k0);
        v16h a1 = ldfrag(a1p + k0);
        v16h b0 = ldfrag(bp + k0);
        v16h b1 = ldfrag(bp + bs + k0);
        v16h b2 = ldfrag(bp + 2 * bs + k0);
        v16h b3 = ldfrag(bp + 3 * bs + k0);
        acc[0] = __builtin_amdgcn_wmma_f32_16x16x32_f16(false, a0, false, b0, (short)0, acc[0], false, false);
        acc[1] = __builtin_amdgcn_wmma_f32_16x16x32_f16(false, a0, false, b1, (short)0, acc[1], false, false);
        acc[2] = __builtin_amdgcn_wmma_f32_16x16x32_f16(false, a0, false, b2, (short)0, acc[2], false, false);
        acc[3] = __builtin_amdgcn_wmma_f32_16x16x32_f16(false, a0, false, b3, (short)0, acc[3], false, false);
        acc[4] = __builtin_amdgcn_wmma_f32_16x16x32_f16(false, a1, false, b0, (short)0, acc[4], false, false);
        acc[5] = __builtin_amdgcn_wmma_f32_16x16x32_f16(false, a1, false, b1, (short)0, acc[5], false, false);
        acc[6] = __builtin_amdgcn_wmma_f32_16x16x32_f16(false, a1, false, b2, (short)0, acc[6], false, false);
        acc[7] = __builtin_amdgcn_wmma_f32_16x16x32_f16(false, a1, false, b3, (short)0, acc[7], false, false);
    }

#pragma unroll
    for (int s = 0; s < 4; s++) {
        int col = nCol + s * 16 + l15;
        if (col < N) {                                // per-lane store guard (after all WMMA)
            float bv = bias ? bias[col] : 0.0f;
#pragma unroll
            for (int r = 0; r < 8; r++) {
                int m0 = mRow + r + half * 8;
                size_t o0 = (size_t)m0 * N + col;
                float v0 = acc[s][r] + bv;
                if (accFlag) v0 += C[o0];
                C[o0] = v0;
                size_t o1 = (size_t)(m0 + 16) * N + col;
                float v1 = acc[4 + s][r] + bv;
                if (accFlag) v1 += C[o1];
                C[o1] = v1;
            }
        }
    }
}

// ---------------- conversion / gather kernels ----------------
__global__ void k_w_transpose_f16(const float* __restrict__ W, h16* __restrict__ Wt,
                                  int K, int N) {
    long i = (long)blockIdx.x * 256 + threadIdx.x;
    if (i >= (long)K * N) return;
    int n = (int)(i / K), k = (int)(i % K);
    Wt[i] = (h16)W[(size_t)k * N + n];
}

__global__ void k_f32_to_f16(const float* __restrict__ s, h16* __restrict__ d, long n) {
    long i = (long)blockIdx.x * 256 + threadIdx.x;
    if (i < n) d[i] = (h16)s[i];
}

__global__ void k_zero_f32(float* p, long n) {
    long i = (long)blockIdx.x * 256 + threadIdx.x;
    if (i < n) p[i] = 0.0f;
}

__global__ void k_zero_f16(h16* p, long n) {
    long i = (long)blockIdx.x * 256 + threadIdx.x;
    if (i < n) p[i] = (h16)0.0f;
}

// x[b*S+s, t] = src_embed[tok]*sqrt(H) + PE(s,t)
__global__ void k_embed_src(const float* __restrict__ emb, const int* __restrict__ src,
                            float* __restrict__ x) {
    int rs = blockIdx.x;             // 0..BS-1
    int t  = threadIdx.x;            // 0..H-1
    int tok = src[rs];
    int s = rs & (Sc - 1);
    int i2 = t >> 1;
    float div = expf((float)(2 * i2) * (-logf(10000.0f) / (float)Hc));
    float ang = (float)s * div;
    float pe = (t & 1) ? cosf(ang) : sinf(ang);
    x[(size_t)rs * Hc + t] = emb[(size_t)tok * Hc + t] * 16.0f + pe;
}

// dst[(b*n+j)*H+t] = label_embed[toks[b*NPT+lo+j]][t]  (as f16)
__global__ void k_embed_labels_f16(const float* __restrict__ emb, const int* __restrict__ toks,
                                   h16* __restrict__ dst, int n, int lo, long total) {
    long i = (long)blockIdx.x * 256 + threadIdx.x;
    if (i >= total) return;
    int t = (int)(i & (Hc - 1));
    long r = i >> 8;
    int j = (int)(r % n);
    int b = (int)(r / n);
    int tok = toks[(size_t)b * NPTc + lo + j];
    dst[i] = (h16)emb[(size_t)tok * Hc + t];
}

// generic batched contiguous row gather -> f16
__global__ void k_rows_f16(const float* __restrict__ src, h16* __restrict__ dst,
                           int rowsPerB, int rowLen, long batchStride, long srcOff,
                           long total) {
    long i = (long)blockIdx.x * 256 + threadIdx.x;
    if (i >= total) return;
    int t = (int)(i % rowLen);
    long r = i / rowLen;
    int j = (int)(r % rowsPerB);
    int b = (int)(r / rowsPerB);
    dst[i] = (h16)src[(size_t)b * batchStride + srcOff + (size_t)j * rowLen + t];
}

// decoder parent repeat gather -> f16 : dst[(b*n+j)*H+t] = hd[b, plo + j/2, t]
__global__ void k_gather_parent_f16(const float* __restrict__ hd, h16* __restrict__ dst,
                                    int n, int plo, long total) {
    long i = (long)blockIdx.x * 256 + threadIdx.x;
    if (i >= total) return;
    int t = (int)(i & (Hc - 1));
    long r = i >> 8;
    int j = (int)(r % n);
    int b = (int)(r / n);
    dst[i] = (h16)hd[((size_t)b * NPTc + plo + (j >> 1)) * Hc + t];
}

// ---------------- attention (per b,head block; softmax in regs) ----------------
__global__ void k_attention(const float* __restrict__ qb, const float* __restrict__ kb,
                            const float* __restrict__ vb, const int* __restrict__ src,
                            float* __restrict__ aoB) {
    int b  = blockIdx.x / NHEADc;
    int hh = blockIdx.x % NHEADc;
    int qi = threadIdx.x;                 // 0..63
    float qreg[DHc];
#pragma unroll
    for (int d = 0; d < DHc; d++)
        qreg[d] = qb[((size_t)b * Sc + qi) * Hc + hh * DHc + d];
    float scores[Sc];
    float mx = -3.4e38f;
    const float scale = 0.17677669529663687f;   // 1/sqrt(32)
    for (int t = 0; t < Sc; t++) {
        float s = 0.f;
#pragma unroll
        for (int d = 0; d < DHc; d++)
            s += qreg[d] * kb[((size_t)b * Sc + t) * Hc + hh * DHc + d];
        s *= scale;
        if (src[b * Sc + t] == 0) s = -1.0e9f;
        scores[t] = s;
        mx = fmaxf(mx, s);
    }
    float sum = 0.f;
    for (int t = 0; t < Sc; t++) { float e = expf(scores[t] - mx); scores[t] = e; sum += e; }
    float inv = 1.0f / sum;
#pragma unroll
    for (int d = 0; d < DHc; d++) {
        float o = 0.f;
        for (int t = 0; t < Sc; t++)
            o += scores[t] * vb[((size_t)b * Sc + t) * Hc + hh * DHc + d];
        aoB[((size_t)b * Sc + qi) * Hc + hh * DHc + d] = o * inv;
    }
}

// ---------------- layernorm (in place): x = LN(x + resid)*g + b ----------------
__global__ void k_ln(float* __restrict__ x, const float* __restrict__ resid,
                     const float* __restrict__ g, const float* __restrict__ bb) {
    __shared__ float red[Hc];
    int row = blockIdx.x, t = threadIdx.x;
    size_t o = (size_t)row * Hc + t;
    float v = x[o];
    if (resid) v += resid[o];
    red[t] = v; __syncthreads();
    for (int s = Hc / 2; s > 0; s >>= 1) { if (t < s) red[t] += red[t + s]; __syncthreads(); }
    float mean = red[0] * (1.0f / Hc); __syncthreads();
    float d = v - mean;
    red[t] = d * d; __syncthreads();
    for (int s = Hc / 2; s > 0; s >>= 1) { if (t < s) red[t] += red[t + s]; __syncthreads(); }
    float var = red[0] * (1.0f / Hc);
    x[o] = d * rsqrtf(var + 1e-5f) * g[t] + bb[t];
}

__global__ void k_relu(float* p, long n) {
    long i = (long)blockIdx.x * 256 + threadIdx.x;
    if (i < n) p[i] = fmaxf(p[i], 0.0f);
}

// masked pool over PAD positions (faithful to reference: m = (src==0))
__global__ void k_pool(const float* __restrict__ x, const int* __restrict__ src,
                       float* __restrict__ pool) {
    int b = blockIdx.x, t = threadIdx.x;
    float s = 0.f, ms = 0.f;
    for (int j = 0; j < Sc; j++) {
        float m = (src[b * Sc + j] == 0) ? 1.0f : 0.0f;
        s += x[((size_t)b * Sc + j) * Hc + t] * m;
        ms += m;
    }
    pool[b * Hc + t] = s / (ms + 1e-7f);
}

// ---------------- TreeLSTM gates ----------------
__global__ void k_gates_enc(const float* __restrict__ iou, const float* __restrict__ f,
                            float* __restrict__ h, float* __restrict__ c,
                            int n, int lo, int clo, int leaf, long total) {
    long i = (long)blockIdx.x * 256 + threadIdx.x;
    if (i >= total) return;
    int t = (int)(i & (Hc - 1));
    long r = i >> 8;
    int j = (int)(r % n);
    int b = (int)(r / n);
    const float* io = iou + (size_t)r * (3 * Hc);
    float iv = io[t], ov = io[Hc + t], uv = io[2 * Hc + t];
    float cagg = 0.f;
    if (!leaf) {
        const float* fr = f + (size_t)r * (2 * Hc);
        size_t cb = ((size_t)b * NPTc + clo + 2 * j) * Hc + t;
        cagg = sigm(fr[t]) * c[cb] + sigm(fr[Hc + t]) * c[cb + Hc];
    }
    float cn = sigm(iv) * tanhf(uv) + cagg;
    size_t o = ((size_t)b * NPTc + lo + j) * Hc + t;
    c[o] = cn;
    h[o] = sigm(ov) * tanhf(cn);
}

__global__ void k_gates_dec(const float* __restrict__ iou, const float* __restrict__ f,
                            float* __restrict__ hd, float* __restrict__ cd,
                            int n, int lo, int plo, long total) {
    long i = (long)blockIdx.x * 256 + threadIdx.x;
    if (i >= total) return;
    int t = (int)(i & (Hc - 1));
    long r = i >> 8;
    int j = (int)(r % n);
    int b = (int)(r / n);
    const float* io = iou + (size_t)r * (3 * Hc);
    float iv = io[t], ov = io[Hc + t], uv = io[2 * Hc + t];
    float cp = cd[((size_t)b * NPTc + plo + (j >> 1)) * Hc + t];
    float cagg = sigm(f[(size_t)r * Hc + t]) * cp;
    float cn = sigm(iv) * tanhf(uv) + cagg;
    size_t o = ((size_t)b * NPTc + lo + j) * Hc + t;
    cd[o] = cn;
    hd[o] = sigm(ov) * tanhf(cn);
}

// ---------------- bottleneck (tiny ops) ----------------
__global__ void k_sh_bits(const float* __restrict__ hid, const float* __restrict__ W2,
                          const float* __restrict__ b2, float* __restrict__ bits) {
    int i = blockIdx.x * 256 + threadIdx.x;
    if (i >= Bc * 5) return;
    int b = i / 5, j = i % 5;
    float s = b2[j];
    for (int k = 0; k < Hc; k++) s += hid[b * Hc + k] * W2[k * 5 + j];
    bits[i] = (s > 0.0f) ? 1.0f : 0.0f;   // forward value of straight-through bits
}

__global__ void k_bits_expand(const float* __restrict__ bits, const float* __restrict__ W3,
                              const float* __restrict__ b3, float* __restrict__ t1) {
    int b = blockIdx.x, t = threadIdx.x;
    float s = b3[t];
#pragma unroll
    for (int j = 0; j < 5; j++) s += bits[b * 5 + j] * W3[j * Hc + t];
    t1[b * Hc + t] = fmaxf(s, 0.0f);
}

__global__ void k_add_pool_scatter(const float* __restrict__ pbuf, const float* __restrict__ pool,
                                   float* __restrict__ hd) {
    int b = blockIdx.x, t = threadIdx.x;
    hd[(size_t)b * NPTc * Hc + t] = pbuf[b * Hc + t] + pool[b * Hc + t];
}

// ---------------- loss / accuracy over a logits chunk ----------------
__global__ void k_loss(const float* __restrict__ L, const int* __restrict__ y,
                       int rows, int rowStart, float* __restrict__ accum) {
    __shared__ float sv[256];
    __shared__ int   si[256];
    int row = blockIdx.x;
    if (row >= rows) return;
    const float* l = L + (size_t)row * LBLc;
    int t = threadIdx.x;
    float bm = -3.4e38f; int bi = 0;
    for (int j = t; j < LBLc; j += 256) { float v = l[j]; if (v > bm) { bm = v; bi = j; } }
    sv[t] = bm; si[t] = bi; __syncthreads();
    for (int s = 128; s > 0; s >>= 1) {
        if (t < s) {
            if (sv[t + s] > sv[t] || (sv[t + s] == sv[t] && si[t + s] < si[t])) {
                sv[t] = sv[t + s]; si[t] = si[t + s];
            }
        }
        __syncthreads();
    }
    float mx = sv[0]; int amax = si[0]; __syncthreads();
    float se = 0.f;
    for (int j = t; j < LBLc; j += 256) se += expf(l[j] - mx);
    sv[t] = se; __syncthreads();
    for (int s = 128; s > 0; s >>= 1) { if (t < s) sv[t] += sv[t + s]; __syncthreads(); }
    if (t == 0) {
        int lbl = y[rowStart + row];
        float logp = l[lbl] - mx - logf(sv[0]);
        atomicAdd(&accum[0], -logp);
        atomicAdd(&accum[1], (amax == lbl) ? 1.0f : 0.0f);
    }
}

__global__ void k_finalize(const float* __restrict__ accum, float* __restrict__ out) {
    out[0] = accum[0] * (1.0f / (float)NNc);
    out[1] = accum[1] * (1.0f / (float)NNc);
}

// =====================================================================
// host orchestration
// =====================================================================
static inline dim3 g1(long n) { return dim3((unsigned)((n + 255) / 256)); }

static void gemm(hipStream_t st, const h16* A, const h16* Bt, const float* bias,
                 float* C, int M, int N, int K, int acc) {
    // M % 64 == 0 for all call sites; Bt padded to ceil(N/256)*256 rows
    dim3 grid((unsigned)((N + 255) / 256), (unsigned)(M / 64));
    k_wmma_gemm<<<grid, 256, 0, st>>>(A, Bt, bias, C, M, N, K, acc);
}

extern "C" void kernel_launch(void* const* d_in, const int* in_sizes, int n_in,
                              void* d_out, int out_size, void* d_ws, size_t ws_size,
                              hipStream_t stream) {
    (void)in_sizes; (void)n_in; (void)out_size; (void)ws_size;
    hipStream_t st = stream;

    // ---- inputs (setup_inputs dict insertion order; params flattened first) ----
    const float* P_semb = (const float*)d_in[0];
    const float* P_Wq   = (const float*)d_in[1];
    const float* P_bq   = (const float*)d_in[2];
    const float* P_Wk   = (const float*)d_in[3];
    const float* P_bk   = (const float*)d_in[4];
    const float* P_Wv   = (const float*)d_in[5];
    const float* P_bv   = (const float*)d_in[6];
    const float* P_Wo   = (const float*)d_in[7];
    const float* P_bo   = (const float*)d_in[8];
    const float* P_l1g  = (const float*)d_in[9];
    const float* P_l1b  = (const float*)d_in[10];
    const float* P_l2g  = (const float*)d_in[11];
    const float* P_l2b  = (const float*)d_in[12];
    const float* P_W1   = (const float*)d_in[13];
    const float* P_b1   = (const float*)d_in[14];
    const float* P_W2   = (const float*)d_in[15];
    const float* P_b2   = (const float*)d_in[16];
    const float* P_lfg  = (const float*)d_in[17];
    const float* P_lfb  = (const float*)d_in[18];
    const float* P_lemb = (const float*)d_in[19];
    const float* P_eWiou= (const float*)d_in[20];
    const float* P_eUiou= (const float*)d_in[21];
    const float* P_ebiou= (const float*)d_in[22];
    const float* P_eUf  = (const float*)d_in[23];
    const float* P_ebf  = (const float*)d_in[24];
    const float* P_dWiou= (const float*)d_in[25];
    const float* P_dUiou= (const float*)d_in[26];
    const float* P_dbiou= (const float*)d_in[27];
    const float* P_dUf  = (const float*)d_in[28];
    const float* P_dbf  = (const float*)d_in[29];
    const float* P_shW1 = (const float*)d_in[30];
    const float* P_shb1 = (const float*)d_in[31];
    const float* P_shW2 = (const float*)d_in[32];
    const float* P_shb2 = (const float*)d_in[33];
    const float* P_shW3 = (const float*)d_in[34];
    const float* P_shb3 = (const float*)d_in[35];
    const float* P_shW4 = (const float*)d_in[36];
    const float* P_shb4 = (const float*)d_in[37];
    const float* P_logW = (const float*)d_in[38];
    const float* P_logb = (const float*)d_in[39];
    const int*   P_src  = (const int*)d_in[40];
    const int*   P_encx = (const int*)d_in[41];
    const int*   P_decx = (const int*)d_in[42];
    const int*   P_y    = (const int*)d_in[43];
    float* out = (float*)d_out;
    char*  ws  = (char*)d_ws;

    // ---- workspace layout ----
    size_t off = 0;
    auto alloc = [&](size_t bytes) -> size_t {
        off = (off + 255) & ~(size_t)255;
        size_t o = off; off += bytes; return o;
    };
    const size_t WPOOL_H = 4325376;   // f16 weight pool (logW padded to 2048 rows)
    const size_t STAGE_H = 4194304;   // max M*K halves for any GEMM operand
    size_t o_wh  = alloc(WPOOL_H * sizeof(h16));
    size_t o_Ab  = alloc(STAGE_H * sizeof(h16));
    size_t o_Hb  = alloc(STAGE_H * sizeof(h16));
    size_t o_x   = alloc((size_t)BSc * Hc * 4);
    size_t o_qb  = alloc((size_t)BSc * Hc * 4);
    size_t o_kb  = alloc((size_t)BSc * Hc * 4);
    size_t o_vb  = alloc((size_t)BSc * Hc * 4);
    size_t o_aoB = alloc((size_t)BSc * Hc * 4);
    size_t o_aoP = alloc((size_t)BSc * Hc * 4);
    size_t o_ffh = alloc((size_t)BSc * FFc * 4);
    size_t o_pool= alloc((size_t)Bc * Hc * 4);
    size_t o_h   = alloc((size_t)NNc * Hc * 4);
    size_t o_c   = alloc((size_t)NNc * Hc * 4);
    size_t o_iou = alloc((size_t)Bc * 256 * (3 * Hc) * 4);
    size_t o_f   = alloc((size_t)4194304 * 4);
    size_t o_hid = alloc((size_t)Bc * Hc * 4);
    size_t o_bits= alloc((size_t)Bc * 8 * 4);
    size_t o_t1  = alloc((size_t)Bc * Hc * 4);
    size_t o_pb  = alloc((size_t)Bc * Hc * 4);
    size_t o_acc = alloc(256);
    size_t o_logits = o_qb;   // alias: logits chunk (32.8MB) reuses qb..ffh region (~38MB)

    h16* wpool = (h16*)(ws + o_wh);
    h16* wQ    = wpool;                 // 3*65536
    h16* wK    = wQ + 196608;
    h16* wV    = wK + 196608;
    h16* wO    = wV + 196608;
    h16* w1    = wO + 196608;           // 3*262144
    h16* w2    = w1 + 786432;
    h16* eWiou = w2 + 786432;           // 196608
    h16* eUiou = eWiou + 196608;        // 393216
    h16* eUf   = eUiou + 393216;        // 262144
    h16* dWiou = eUf + 262144;          // 196608
    h16* dUiou = dWiou + 196608;        // 196608
    h16* dUf   = dUiou + 196608;        // 65536
    h16* shW1h = dUf + 65536;
    h16* shW4h = shW1h + 65536;
    h16* logWh = shW4h + 65536;         // 2048*256 (rows 2000..2047 zeroed)

    h16*   Ab   = (h16*)(ws + o_Ab);
    h16*   Hb   = (h16*)(ws + o_Hb);
    float* x    = (float*)(ws + o_x);
    float* qb   = (float*)(ws + o_qb);
    float* kb   = (float*)(ws + o_kb);
    float* vb   = (float*)(ws + o_vb);
    float* aoB  = (float*)(ws + o_aoB);
    float* aoP  = (float*)(ws + o_aoP);
    float* ffh  = (float*)(ws + o_ffh);
    float* pool = (float*)(ws + o_pool);
    float* h    = (float*)(ws + o_h);    // reused as hd in decoder
    float* c    = (float*)(ws + o_c);    // reused as cd in decoder
    float* iou  = (float*)(ws + o_iou);
    float* fb   = (float*)(ws + o_f);
    float* hid  = (float*)(ws + o_hid);
    float* bits = (float*)(ws + o_bits);
    float* t1   = (float*)(ws + o_t1);
    float* pb   = (float*)(ws + o_pb);
    float* acc  = (float*)(ws + o_acc);
    float* logits = (float*)(ws + o_logits);

    // ---- convert weights to pre-transposed f16 [N,K] ----
    auto WT = [&](const float* W, h16* Wt, int K, int N) {
        k_w_transpose_f16<<<g1((long)K * N), 256, 0, st>>>(W, Wt, K, N);
    };
    for (int li = 0; li < NLAY; li++) {
        WT(P_Wq + (size_t)li * Hc * Hc, wQ + (size_t)li * Hc * Hc, Hc, Hc);
        WT(P_Wk + (size_t)li * Hc * Hc, wK + (size_t)li * Hc * Hc, Hc, Hc);
        WT(P_Wv + (size_t)li * Hc * Hc, wV + (size_t)li * Hc * Hc, Hc, Hc);
        WT(P_Wo + (size_t)li * Hc * Hc, wO + (size_t)li * Hc * Hc, Hc, Hc);
        WT(P_W1 + (size_t)li * Hc * FFc, w1 + (size_t)li * Hc * FFc, Hc, FFc);
        WT(P_W2 + (size_t)li * FFc * Hc, w2 + (size_t)li * FFc * Hc, FFc, Hc);
    }
    WT(P_eWiou, eWiou, Hc, 3 * Hc);
    WT(P_eUiou, eUiou, 2 * Hc, 3 * Hc);
    WT(P_eUf,   eUf,   2 * Hc, 2 * Hc);
    WT(P_dWiou, dWiou, Hc, 3 * Hc);
    WT(P_dUiou, dUiou, Hc, 3 * Hc);
    WT(P_dUf,   dUf,   Hc, Hc);
    WT(P_shW1,  shW1h, Hc, Hc);
    WT(P_shW4,  shW4h, Hc, Hc);
    WT(P_logW,  logWh, Hc, LBLc);
    // zero the padded logit-weight rows 2000..2047 so B-frag loads are safe
    k_zero_f16<<<g1((long)(LBLpad - LBLc) * Hc), 256, 0, st>>>(
        logWh + (size_t)LBLc * Hc, (long)(LBLpad - LBLc) * Hc);

    // ================= phase 1: transformer encoder =================
    k_embed_src<<<BSc, Hc, 0, st>>>(P_semb, P_src, x);
    for (int li = 0; li < NLAY; li++) {
        size_t wo = (size_t)li * Hc * Hc;
        k_f32_to_f16<<<g1((long)BSc * Hc), 256, 0, st>>>(x, Ab, (long)BSc * Hc);
        gemm(st, Ab, wQ + wo, P_bq + li * Hc, qb, BSc, Hc, Hc, 0);
        gemm(st, Ab, wK + wo, P_bk + li * Hc, kb, BSc, Hc, Hc, 0);
        gemm(st, Ab, wV + wo, P_bv + li * Hc, vb, BSc, Hc, Hc, 0);
        k_attention<<<Bc * NHEADc, Sc, 0, st>>>(qb, kb, vb, P_src, aoB);
        k_f32_to_f16<<<g1((long)BSc * Hc), 256, 0, st>>>(aoB, Hb, (long)BSc * Hc);
        gemm(st, Hb, wO + wo, P_bo + li * Hc, aoP, BSc, Hc, Hc, 0);
        k_ln<<<BSc, Hc, 0, st>>>(x, aoP, P_l1g + li * Hc, P_l1b + li * Hc);
        k_f32_to_f16<<<g1((long)BSc * Hc), 256, 0, st>>>(x, Ab, (long)BSc * Hc);
        gemm(st, Ab, w1 + (size_t)li * Hc * FFc, P_b1 + li * FFc, ffh, BSc, FFc, Hc, 0);
        k_relu<<<g1((long)BSc * FFc), 256, 0, st>>>(ffh, (long)BSc * FFc);
        k_f32_to_f16<<<g1((long)BSc * FFc), 256, 0, st>>>(ffh, Ab, (long)BSc * FFc);
        gemm(st, Ab, w2 + (size_t)li * FFc * Hc, P_b2 + li * Hc, aoP, BSc, Hc, FFc, 0);
        k_ln<<<BSc, Hc, 0, st>>>(x, aoP, P_l2g + li * Hc, P_l2b + li * Hc);
    }
    k_ln<<<BSc, Hc, 0, st>>>(x, nullptr, P_lfg, P_lfb);
    k_pool<<<Bc, Hc, 0, st>>>(x, P_src, pool);

    // ================= phase 2: TreeLSTM encoder (bottom-up) =================
    for (int l = DEPTHc - 1; l >= 0; l--) {
        int n = 1 << l;
        int lo = (1 << l) - 1;
        int clo = (1 << (l + 1)) - 1;
        long rowsT = (long)Bc * n;
        k_embed_labels_f16<<<g1(rowsT * Hc), 256, 0, st>>>(P_lemb, P_encx, Ab, n, lo, rowsT * Hc);
        gemm(st, Ab, eWiou, P_ebiou, iou, (int)rowsT, 3 * Hc, Hc, 0);
        if (l < DEPTHc - 1) {
            k_rows_f16<<<g1(rowsT * 2 * Hc), 256, 0, st>>>(h, Hb, n, 2 * Hc,
                                                           (long)NPTc * Hc, (long)clo * Hc,
                                                           rowsT * 2 * Hc);
            gemm(st, Hb, eUf, P_ebf, fb, (int)rowsT, 2 * Hc, 2 * Hc, 0);
            gemm(st, Hb, eUiou, nullptr, iou, (int)rowsT, 3 * Hc, 2 * Hc, 1);
        }
        k_gates_enc<<<g1(rowsT * Hc), 256, 0, st>>>(iou, fb, h, c, n, lo, clo,
                                                    (l == DEPTHc - 1) ? 1 : 0, rowsT * Hc);
    }

    // ================= phase 3: bottleneck =================
    k_rows_f16<<<g1((long)Bc * Hc), 256, 0, st>>>(h, Ab, 1, Hc, (long)NPTc * Hc, 0,
                                                  (long)Bc * Hc);          // root_h -> f16
    gemm(st, Ab, shW1h, P_shb1, hid, Bc, Hc, Hc, 0);
    k_relu<<<g1((long)Bc * Hc), 256, 0, st>>>(hid, (long)Bc * Hc);
    k_sh_bits<<<2, 256, 0, st>>>(hid, P_shW2, P_shb2, bits);
    k_bits_expand<<<Bc, Hc, 0, st>>>(bits, P_shW3, P_shb3, t1);
    k_f32_to_f16<<<g1((long)Bc * Hc), 256, 0, st>>>(t1, Ab, (long)Bc * Hc);
    gemm(st, Ab, shW4h, P_shb4, pb, Bc, Hc, Hc, 0);
    k_zero_f32<<<g1((long)NNc * Hc), 256, 0, st>>>(c, (long)NNc * Hc);     // cd := 0
    k_add_pool_scatter<<<Bc, Hc, 0, st>>>(pb, pool, h);                    // hd[:,0] = partial_h

    // ================= phase 4: TreeLSTM decoder (top-down) =================
    for (int l = 1; l < DEPTHc; l++) {
        int n = 1 << l;
        int lo = (1 << l) - 1;
        int plo = (1 << (l - 1)) - 1;
        long rowsT = (long)Bc * n;
        k_embed_labels_f16<<<g1(rowsT * Hc), 256, 0, st>>>(P_lemb, P_decx, Ab, n, lo, rowsT * Hc);
        gemm(st, Ab, dWiou, P_dbiou, iou, (int)rowsT, 3 * Hc, Hc, 0);
        k_gather_parent_f16<<<g1(rowsT * Hc), 256, 0, st>>>(h, Hb, n, plo, rowsT * Hc);
        gemm(st, Hb, dUf, P_dbf, fb, (int)rowsT, Hc, Hc, 0);
        gemm(st, Hb, dUiou, nullptr, iou, (int)rowsT, 3 * Hc, Hc, 1);
        k_gates_dec<<<g1(rowsT * Hc), 256, 0, st>>>(iou, fb, h, c, n, lo, plo, rowsT * Hc);
    }

    // ================= phase 5: logits + loss (chunked) =================
    k_zero_f32<<<1, 256, 0, st>>>(acc, 2);
    const int CHUNK = 4096;
    for (int start = 0; start < NNc; start += CHUNK) {
        int rows = (NNc - start < CHUNK) ? (NNc - start) : CHUNK;   // 4096 ... 4032 (both %64==0)
        k_f32_to_f16<<<g1((long)rows * Hc), 256, 0, st>>>(h + (size_t)start * Hc, Ab,
                                                          (long)rows * Hc);
        gemm(st, Ab, logWh, P_logb, logits, rows, LBLc, Hc, 0);
        k_loss<<<rows, 256, 0, st>>>(logits, P_y, rows, start, acc);
    }
    k_finalize<<<1, 1, 0, st>>>(acc, out);
}
